// ChunkedDeltaNetLayer_62302795596469
// MI455X (gfx1250) — compile-verified, matrix operand order checked by reading, not run on hardware
//
#include <hip/hip_runtime.h>
#include <cstdint>
#include <cstddef>

// ---------------------------------------------------------------------------
// DeltaNet layer forward for MI455X (gfx1250, wave32, WMMA 16x16x32 bf16).
//   x(B,S,1024) -> q,k,v = act(x@W^T); chunked delta-rule scan per head with
//   64x64 f32 state in LDS; RMSNorm; out = normed @ Wo^T.
// Big GEMMs + all chunk-level GEMMs run on v_wmma_f32_16x16x32_bf16 (bf16 in,
// f32 accumulate). Global->LDS staging uses gfx1250 async copies
// (global_load_async_to_lds_b128, ASYNCcnt) with double buffering, plus
// global_prefetch via __builtin_prefetch. Sequence-serial part is reduced to
// a CxC forward substitution per chunk (exact f32); everything else is WMMA.
// ---------------------------------------------------------------------------

#define DEV __device__ __forceinline__

typedef __attribute__((ext_vector_type(16))) __bf16 v16bf;
typedef __attribute__((ext_vector_type(8)))  float  v8f;

namespace dn {
constexpr int B  = 2;
constexpr int S  = 1024;
constexpr int DM = 1024;
constexpr int H  = 16;
constexpr int HD = 64;           // head dim
constexpr int C  = 32;           // scan chunk length
constexpr int NC = S / C;        // chunks per sequence
}

// ------------------- CDNA5 async global->LDS copy helpers -------------------
// Generic LDS pointer = {shared aperture, 32-bit 0-based LDS offset}; the
// async instructions take that 32-bit offset in the VDST VGPR (ISA ch.10.7).
DEV uint32_t lds_off32(const void* p) { return (uint32_t)(uintptr_t)p; }

DEV void async_load_b128(uint32_t lds, const void* g) {
  asm volatile("global_load_async_to_lds_b128 %0, %1, off"
               :: "v"(lds), "v"((unsigned long long)(uintptr_t)g)
               : "memory");
}
DEV void wait_async0() { asm volatile("s_wait_asynccnt 0x0" ::: "memory"); }

// ---------------- WMMA fragment helpers (ISA 7.12.2, wave32) ----------------
// 16-bit A 16x32: lane m<16 holds row m, K {0..7,16..23}; lane m+16 holds row m,
// K {8..15,24..31}. v16bf element f -> VGPR f/2, half f%2.
DEV int frag_k_of(int j, int hi) {            // even K for vgpr j, half-group hi
  return ((j & 3) * 2) + hi * 8 + ((j >= 4) ? 16 : 0);
}

// A fragment from row-major (m,k) tile, 16 rows x 32 k, row stride ld.
DEV v16bf frag_a_bf(const __bf16* p, int ld, int lane) {
  const int m = lane & 15, hi = lane >> 4;
  v16bf f;
#pragma unroll
  for (int j = 0; j < 8; ++j) {
    const int k = frag_k_of(j, hi);
    f[2 * j]     = p[m * ld + k];
    f[2 * j + 1] = p[m * ld + k + 1];
  }
  return f;
}
DEV v16bf frag_a_f32(const float* p, int ld, int lane) {
  const int m = lane & 15, hi = lane >> 4;
  v16bf f;
#pragma unroll
  for (int j = 0; j < 8; ++j) {
    const int k = frag_k_of(j, hi);
    f[2 * j]     = (__bf16)p[m * ld + k];
    f[2 * j + 1] = (__bf16)p[m * ld + k + 1];
  }
  return f;
}

// B fragment where B[k][n] = p[n*ld + k]  (storage is (n,k) row-major,
// the natural layout for W stored as (N,K) and transposed-operand GEMMs).
DEV v16bf frag_b_nk_bf(const __bf16* p, int ld, int lane) {
  const int n = lane & 15, hi = lane >> 4;
  v16bf f;
#pragma unroll
  for (int j = 0; j < 8; ++j) {
    const int k = frag_k_of(j, hi);
    f[2 * j]     = p[n * ld + k];
    f[2 * j + 1] = p[n * ld + k + 1];
  }
  return f;
}
DEV v16bf frag_b_nk_f32(const float* p, int ld, int lane) {
  const int n = lane & 15, hi = lane >> 4;
  v16bf f;
#pragma unroll
  for (int j = 0; j < 8; ++j) {
    const int k = frag_k_of(j, hi);
    f[2 * j]     = (__bf16)p[n * ld + k];
    f[2 * j + 1] = (__bf16)p[n * ld + k + 1];
  }
  return f;
}

// B fragment where B[k][n] = p[k*ld + n]  (storage is (k,n) row-major).
DEV v16bf frag_b_km_bf(const __bf16* p, int ld, int lane) {
  const int n = lane & 15, hi = lane >> 4;
  v16bf f;
#pragma unroll
  for (int j = 0; j < 8; ++j) {
    const int k = frag_k_of(j, hi);
    f[2 * j]     = p[k * ld + n];
    f[2 * j + 1] = p[(k + 1) * ld + n];
  }
  return f;
}
DEV v16bf frag_b_km_f32(const float* p, int ld, int lane) {
  const int n = lane & 15, hi = lane >> 4;
  v16bf f;
#pragma unroll
  for (int j = 0; j < 8; ++j) {
    const int k = frag_k_of(j, hi);
    f[2 * j]     = (__bf16)p[k * ld + n];
    f[2 * j + 1] = (__bf16)p[(k + 1) * ld + n];
  }
  return f;
}

// C/D f32 16x16: VGPR r -> row r + 8*hi, col = lane&15.
DEV void store_c_f32(float* p, int ld, int lane, v8f c) {
  const int n = lane & 15, hi = lane >> 4;
#pragma unroll
  for (int r = 0; r < 8; ++r) p[(r + 8 * hi) * ld + n] = c[r];
}
DEV v8f load_c_f32(const float* p, int ld, int lane) {
  const int n = lane & 15, hi = lane >> 4;
  v8f c;
#pragma unroll
  for (int r = 0; r < 8; ++r) c[r] = p[(r + 8 * hi) * ld + n];
  return c;
}

DEV v8f wmma_bf16(v16bf a, v16bf b, v8f c) {
  return __builtin_amdgcn_wmma_f32_16x16x32_bf16(false, a, false, b, (short)0, c,
                                                 false, false);
}

DEV float wave_sum32(float v) {
#pragma unroll
  for (int m = 16; m >= 1; m >>= 1) v += __shfl_xor(v, m, 32);
  return v;
}

DEV float sigmoidf_(float x) { return 1.f / (1.f + expf(-x)); }

// ---------------------------- f32 -> bf16 convert ---------------------------
__global__ void __launch_bounds__(256) cvt_bf16(const float* __restrict__ src,
                                                __bf16* __restrict__ dst, int n) {
  int i = blockIdx.x * 256 + threadIdx.x;
  if (i < n) dst[i] = (__bf16)src[i];
}

// ------------------- bf16 GEMM: C[m,n] = sum_k A[m,k] W[n,k] ----------------
// A (M,K) row-major bf16, W (N,K) row-major bf16 (computes A @ W^T),
// C (M,N) f32. Block = 128 threads (4 waves); block tile 64x64; K step 32.
// Double-buffered LDS fed by async global->LDS copies; L2 prefetch 2 steps out.
__global__ void __launch_bounds__(128) gemm_bf16_nt(const __bf16* __restrict__ A,
                                                    const __bf16* __restrict__ W,
                                                    float* __restrict__ Cout,
                                                    int M, int N, int K) {
  __shared__ __align__(16) __bf16 sA[2][64 * 32];
  __shared__ __align__(16) __bf16 sW[2][64 * 32];
  const int m0 = blockIdx.x * 64, n0 = blockIdx.y * 64;
  const int tid = threadIdx.x, lane = tid & 31, wv = tid >> 5;

  v8f acc[4] = {v8f{}, v8f{}, v8f{}, v8f{}};

  // one 16B slot = 8 bf16; 256 slots per 64x32 tile; 2 slots/thread/tile
  auto issue = [&](int buf, int k0) {
    for (int i = tid; i < 256; i += 128) {
      const int row = i >> 2, c4 = i & 3;
      async_load_b128(lds_off32(&sA[buf][i * 8]),
                      A + (size_t)(m0 + row) * K + k0 + c4 * 8);
      async_load_b128(lds_off32(&sW[buf][i * 8]),
                      W + (size_t)(n0 + row) * K + k0 + c4 * 8);
    }
  };

  issue(0, 0);
  wait_async0();
  __syncthreads();

  const int nk = K >> 5;
  for (int kc = 0; kc < nk; ++kc) {
    const int cur = kc & 1;
    if (kc + 1 < nk) issue(cur ^ 1, (kc + 1) << 5);
    if (kc + 2 < nk) {
      const int kp = (kc + 2) << 5;
      __builtin_prefetch(A + (size_t)(m0 + (tid >> 2)) * K + kp, 0, 0);
      __builtin_prefetch(W + (size_t)(n0 + (tid >> 2)) * K + kp, 0, 0);
    }
    const v16bf af = frag_a_bf(&sA[cur][(wv * 16) * 32], 32, lane);
#pragma unroll
    for (int tc = 0; tc < 4; ++tc) {
      const v16bf bf = frag_b_nk_bf(&sW[cur][(tc * 16) * 32], 32, lane);
      acc[tc] = wmma_bf16(af, bf, acc[tc]);
    }
    wait_async0();
    __syncthreads();
  }

  const int hi = lane >> 4, col = lane & 15;
#pragma unroll
  for (int tc = 0; tc < 4; ++tc)
#pragma unroll
    for (int r = 0; r < 8; ++r)
      Cout[(size_t)(m0 + wv * 16 + r + 8 * hi) * N + n0 + tc * 16 + col] = acc[tc][r];
}

// ------------------------- beta = 2*sigmoid(x @ Wb^T) -----------------------
__global__ void __launch_bounds__(128) beta_proj(const float* __restrict__ x,
                                                 const float* __restrict__ Wb,
                                                 float* __restrict__ beta) {
  const int bs = blockIdx.x;                       // (b*S + s)
  const int lane = threadIdx.x & 31, wv = threadIdx.x >> 5;
  const int b = bs / dn::S, s = bs % dn::S;
  const float* xr = x + (size_t)bs * dn::DM;
  for (int h = wv; h < dn::H; h += 4) {
    const float* w = Wb + (size_t)h * dn::DM;
    float acc = 0.f;
    for (int k = lane; k < dn::DM; k += 32) acc += xr[k] * w[k];
    acc = wave_sum32(acc);
    if (lane == 0)
      beta[((size_t)(b * dn::H + h)) * dn::S + s] = 2.f * sigmoidf_(acc);
  }
}

// ------- q = l2norm(silu(qraw)), k = l2norm(silu(kraw)), v = silu(vraw) -----
// One wave per (b,s,h) head row of 64; outputs repacked to (b,h,s,d).
__global__ void __launch_bounds__(128) qkv_act(const float* __restrict__ qraw,
                                               const float* __restrict__ kraw,
                                               const float* __restrict__ vraw,
                                               __bf16* __restrict__ qbf,
                                               __bf16* __restrict__ kbf,
                                               float* __restrict__ vbh) {
  const int gh = blockIdx.x * 4 + (threadIdx.x >> 5);   // (b*S+s)*H + h
  const int lane = threadIdx.x & 31;
  const int h = gh & (dn::H - 1);
  const int bs = gh >> 4;
  const int b = bs / dn::S, s = bs % dn::S;
  const size_t src = (size_t)bs * dn::DM + (size_t)h * dn::HD;
  const size_t dst = (((size_t)(b * dn::H + h)) * dn::S + s) * dn::HD;

  {
    float a0 = qraw[src + lane], a1 = qraw[src + 32 + lane];
    a0 *= sigmoidf_(a0); a1 *= sigmoidf_(a1);
    const float ss = wave_sum32(a0 * a0 + a1 * a1);
    const float inv = 1.f / (sqrtf(ss) + 1e-6f);
    qbf[dst + lane]      = (__bf16)(a0 * inv);
    qbf[dst + 32 + lane] = (__bf16)(a1 * inv);
  }
  {
    float a0 = kraw[src + lane], a1 = kraw[src + 32 + lane];
    a0 *= sigmoidf_(a0); a1 *= sigmoidf_(a1);
    const float ss = wave_sum32(a0 * a0 + a1 * a1);
    const float inv = 1.f / (sqrtf(ss) + 1e-6f);
    kbf[dst + lane]      = (__bf16)(a0 * inv);
    kbf[dst + 32 + lane] = (__bf16)(a1 * inv);
  }
  {
    float a0 = vraw[src + lane], a1 = vraw[src + 32 + lane];
    vbh[dst + lane]      = a0 * sigmoidf_(a0);
    vbh[dst + 32 + lane] = a1 * sigmoidf_(a1);
  }
}

// --------------------------- chunked delta-rule scan ------------------------
// Per (b,h): state M (64x64 f32) in LDS. Per chunk (C=32):
//   R0 = K M^T; Bc = V - R0; G = K K^T; P = Bc K^T (WMMA)
//   w[s,t] = beta_s (P[s,t] - sum_{r<s} w[r,s] G[r,t])   (forward subst, f32)
//   Delta = Bc - W^T K ; O = Q M + mask(Q K^T * beta) Delta (WMMA)
//   M += K^T diag(beta) Delta (WMMA)
// Q/K chunk tiles are double-buffered and fetched with async global->LDS
// copies one chunk ahead, hiding global latency behind the chunk's WMMAs.
__global__ void __launch_bounds__(128) delta_scan(const __bf16* __restrict__ qbf,
                                                  const __bf16* __restrict__ kbf,
                                                  const float* __restrict__ vbh,
                                                  const float* __restrict__ beta,
                                                  float* __restrict__ att) {
  using namespace dn;
  __shared__ float  sM[64 * 64];                       // state, f32 (16K)
  __shared__ __align__(16) __bf16 sQ[2][C * 64];       // 2 x 4K
  __shared__ __align__(16) __bf16 sK[2][C * 64];       // 2 x 4K
  __shared__ float  sB[C * 64];        // Bc = V - K M^T (f32, 8K)
  __shared__ __bf16 sD[C * 64];        // Delta (bf16, 4K)
  __shared__ __bf16 sKbT[64 * C];      // KbT[i][t] = beta_t * K[t][i] (4K)
  __shared__ float  sG[C * C];
  __shared__ float  sP[C * C];         // P, later L (f32)
  __shared__ float  sW[C * C];
  __shared__ __bf16 sL[C * C];
  __shared__ float  sbeta[C];

  const int bh = blockIdx.x;           // 0..B*H-1
  const int b = bh >> 4, h = bh & 15;
  const int tid = threadIdx.x, lane = tid & 31, wv = tid >> 5;
  const size_t head = ((size_t)(b * H + h)) * S * HD;

  // async-stage one chunk's Q and K (C*64 bf16 = 256 x 16B slots each)
  auto issue_chunk = [&](int buf, int c) {
    const size_t base = head + (size_t)(c * C) * HD;
    for (int i = tid; i < 256; i += 128) {
      async_load_b128(lds_off32(&sQ[buf][i * 8]), qbf + base + i * 8);
      async_load_b128(lds_off32(&sK[buf][i * 8]), kbf + base + i * 8);
    }
  };

  for (int i = tid; i < 64 * 64; i += 128) sM[i] = 0.f;
  issue_chunk(0, 0);
  wait_async0();
  __syncthreads();

  for (int c = 0; c < NC; ++c) {
    const int cur = c & 1;
    const int s0 = c * C;
    if (c + 1 < NC) issue_chunk(cur ^ 1, c + 1);     // prefetch next chunk
    if (tid < C) sbeta[tid] = beta[((size_t)(b * H + h)) * S + s0 + tid];
    __syncthreads();
    const __bf16* cQ = sQ[cur];
    const __bf16* cK = sK[cur];

    // ---- R0 = K M^T  (C x 64), tiles: tc = wv, tr = 0..1 ----
#pragma unroll
    for (int tr = 0; tr < 2; ++tr) {
      v8f acc = {};
#pragma unroll
      for (int kk = 0; kk < 64; kk += 32) {
        const v16bf a = frag_a_bf(&cK[(tr * 16) * 64 + kk], 64, lane);
        const v16bf bb = frag_b_nk_f32(&sM[(wv * 16) * 64 + kk], 64, lane);
        acc = wmma_bf16(a, bb, acc);
      }
      store_c_f32(&sB[(tr * 16) * 64 + wv * 16], 64, lane, acc);
    }
    __syncthreads();

    // ---- Bc = V - R0 ----
    for (int i = tid; i < C * 64; i += 128)
      sB[i] = vbh[head + (size_t)s0 * HD + i] - sB[i];
    __syncthreads();

    // ---- G = K K^T, P = Bc K^T (each wave one 16x16 tile of each) ----
    {
      const int tr = wv >> 1, tc = wv & 1;
      v8f accG = {}, accP = {};
#pragma unroll
      for (int kk = 0; kk < 64; kk += 32) {
        const v16bf bk = frag_b_nk_bf(&cK[(tc * 16) * 64 + kk], 64, lane);
        const v16bf ak = frag_a_bf(&cK[(tr * 16) * 64 + kk], 64, lane);
        const v16bf ab = frag_a_f32(&sB[(tr * 16) * 64 + kk], 64, lane);
        accG = wmma_bf16(ak, bk, accG);
        accP = wmma_bf16(ab, bk, accP);
      }
      store_c_f32(&sG[(tr * 16) * C + tc * 16], C, lane, accG);
      store_c_f32(&sP[(tr * 16) * C + tc * 16], C, lane, accP);
    }
    __syncthreads();

    // ---- O1 = Q M (accumulators stay in registers for O2) ----
    v8f accO[2];
#pragma unroll
    for (int tr = 0; tr < 2; ++tr) {
      v8f acc = {};
#pragma unroll
      for (int kk = 0; kk < 64; kk += 32) {
        const v16bf a = frag_a_bf(&cQ[(tr * 16) * 64 + kk], 64, lane);
        const v16bf bb = frag_b_km_f32(&sM[kk * 64 + wv * 16], 64, lane);
        acc = wmma_bf16(a, bb, acc);
      }
      accO[tr] = acc;
    }

    // ---- forward substitution for W (exact f32, C iterations) ----
    for (int s = 0; s < C; ++s) {
      if (tid < C) {
        const int t = tid;
        float w = 0.f;
        if (t > s) {
          float d = sP[s * C + t];
          for (int r = 0; r < s; ++r) d -= sW[r * C + s] * sG[r * C + t];
          w = sbeta[s] * d;
        }
        sW[s * C + t] = w;
      }
      __syncthreads();
    }

    // ---- L' = Q K^T into sP (P no longer needed) ----
    {
      const int tr = wv >> 1, tc = wv & 1;
      v8f acc = {};
#pragma unroll
      for (int kk = 0; kk < 64; kk += 32) {
        const v16bf a = frag_a_bf(&cQ[(tr * 16) * 64 + kk], 64, lane);
        const v16bf bb = frag_b_nk_bf(&cK[(tc * 16) * 64 + kk], 64, lane);
        acc = wmma_bf16(a, bb, acc);
      }
      store_c_f32(&sP[(tr * 16) * C + tc * 16], C, lane, acc);
    }
    __syncthreads();

    // ---- Delta = Bc - W^T K ; KbT = diag(beta) K transposed ; mask L ----
    for (int i = tid; i < C * 64; i += 128) {
      const int t = i >> 6, j = i & 63;
      float d = sB[i];
      for (int s = 0; s < t; ++s) d -= sW[s * C + t] * (float)cK[s * 64 + j];
      sD[i] = (__bf16)d;
    }
    for (int i = tid; i < 64 * C; i += 128) {
      const int ii = i >> 5, t = i & (C - 1);
      sKbT[i] = (__bf16)(sbeta[t] * (float)cK[t * 64 + ii]);
    }
    for (int i = tid; i < C * C; i += 128) {
      const int t = i >> 5, s = i & (C - 1);
      sL[i] = (__bf16)((s < t) ? sbeta[s] * sP[i] : 0.f);
    }
    __syncthreads();

    // ---- O += L Delta ; store chunk output to att(b,s,dm) ----
#pragma unroll
    for (int tr = 0; tr < 2; ++tr) {
      const v16bf a = frag_a_bf(&sL[(tr * 16) * C], C, lane);
      const v16bf bb = frag_b_km_bf(&sD[wv * 16], 64, lane);
      accO[tr] = wmma_bf16(a, bb, accO[tr]);
      const int hi = lane >> 4, col = lane & 15;
      const int row0 = s0 + tr * 16, col0 = h * HD + wv * 16;
#pragma unroll
      for (int r = 0; r < 8; ++r)
        att[((size_t)b * S + row0 + r + 8 * hi) * DM + col0 + col] = accO[tr][r];
    }

    // ---- M += K^T diag(beta) Delta  (tc = wv, tr = 0..3) ----
#pragma unroll
    for (int tr = 0; tr < 4; ++tr) {
      v8f cc = load_c_f32(&sM[(tr * 16) * 64 + wv * 16], 64, lane);
      const v16bf a = frag_a_bf(&sKbT[(tr * 16) * C], C, lane);
      const v16bf bb = frag_b_km_bf(&sD[wv * 16], 64, lane);
      cc = wmma_bf16(a, bb, cc);
      store_c_f32(&sM[(tr * 16) * 64 + wv * 16], 64, lane, cc);
    }
    wait_async0();        // next chunk's Q/K async copies have landed
    __syncthreads();
  }
}

// --------------------------------- RMSNorm ----------------------------------
__global__ void __launch_bounds__(256) rmsnorm_bf16(const float* __restrict__ att,
                                                    const float* __restrict__ w,
                                                    __bf16* __restrict__ out) {
  __shared__ float red[8];
  const int bs = blockIdx.x;
  const float* r = att + (size_t)bs * dn::DM;
  float ss = 0.f;
  for (int i = threadIdx.x; i < dn::DM; i += 256) { const float v = r[i]; ss += v * v; }
  ss = wave_sum32(ss);
  const int lane = threadIdx.x & 31, wv = threadIdx.x >> 5;
  if (lane == 0) red[wv] = ss;
  __syncthreads();
  float tot = 0.f;
#pragma unroll
  for (int j = 0; j < 8; ++j) tot += red[j];
  const float inv = 1.f / sqrtf(tot / (float)dn::DM + 1e-6f);
  for (int i = threadIdx.x; i < dn::DM; i += 256)
    out[(size_t)bs * dn::DM + i] = (__bf16)(r[i] * inv * w[i]);
}

// --------------------------------- launcher ---------------------------------
extern "C" void kernel_launch(void* const* d_in, const int* in_sizes, int n_in,
                              void* d_out, int out_size, void* d_ws, size_t ws_size,
                              hipStream_t stream) {
  using namespace dn;
  (void)in_sizes; (void)n_in; (void)out_size; (void)ws_size;
  const float* x    = (const float*)d_in[0];
  const float* Wq   = (const float*)d_in[1];
  const float* Wk   = (const float*)d_in[2];
  const float* Wv   = (const float*)d_in[3];
  const float* Wb   = (const float*)d_in[4];
  const float* Wo   = (const float*)d_in[5];
  const float* rmsw = (const float*)d_in[6];
  float* out = (float*)d_out;

  const int MR = B * S;   // 2048 token rows
  char* ws = (char*)d_ws;
  size_t off = 0;
  auto take = [&](size_t bytes) -> char* {
    char* p = ws + off;
    off += (bytes + 255) & ~(size_t)255;
    return p;
  };
  __bf16* xbf  = (__bf16*)take((size_t)MR * DM * 2);
  __bf16* wqb  = (__bf16*)take((size_t)DM * DM * 2);
  __bf16* wkb  = (__bf16*)take((size_t)DM * DM * 2);
  __bf16* wvb  = (__bf16*)take((size_t)DM * DM * 2);
  __bf16* wob  = (__bf16*)take((size_t)DM * DM * 2);
  float*  qraw = (float*)take((size_t)MR * DM * 4);
  float*  kraw = (float*)take((size_t)MR * DM * 4);
  float*  vraw = (float*)take((size_t)MR * DM * 4);
  __bf16* qbf  = (__bf16*)take((size_t)MR * DM * 2);
  __bf16* kbf  = (__bf16*)take((size_t)MR * DM * 2);
  float*  vbh  = (float*)take((size_t)MR * DM * 4);
  float*  betb = (float*)take((size_t)B * H * S * 4);
  float*  att  = (float*)take((size_t)MR * DM * 4);
  __bf16* nrm  = (__bf16*)take((size_t)MR * DM * 2);

  // f32 -> bf16 operand conversion
  {
    int n = MR * DM;
    cvt_bf16<<<(n + 255) / 256, 256, 0, stream>>>(x, xbf, n);
    n = DM * DM;
    cvt_bf16<<<(n + 255) / 256, 256, 0, stream>>>(Wq, wqb, n);
    cvt_bf16<<<(n + 255) / 256, 256, 0, stream>>>(Wk, wkb, n);
    cvt_bf16<<<(n + 255) / 256, 256, 0, stream>>>(Wv, wvb, n);
    cvt_bf16<<<(n + 255) / 256, 256, 0, stream>>>(Wo, wob, n);
  }

  // projections (WMMA, async-fed)
  dim3 gg(MR / 64, DM / 64);
  gemm_bf16_nt<<<gg, 128, 0, stream>>>(xbf, wqb, qraw, MR, DM, DM);
  gemm_bf16_nt<<<gg, 128, 0, stream>>>(xbf, wkb, kraw, MR, DM, DM);
  gemm_bf16_nt<<<gg, 128, 0, stream>>>(xbf, wvb, vraw, MR, DM, DM);
  beta_proj<<<MR, 128, 0, stream>>>(x, Wb, betb);

  // activations + head repack
  qkv_act<<<MR * H / 4, 128, 0, stream>>>(qraw, kraw, vraw, qbf, kbf, vbh);

  // chunked delta-rule scan (WMMA inside, async chunk prefetch)
  delta_scan<<<B * H, 128, 0, stream>>>(qbf, kbf, vbh, betb, att);

  // RMSNorm then output projection (WMMA)
  rmsnorm_bf16<<<MR, 256, 0, stream>>>(att, rmsw, nrm);
  gemm_bf16_nt<<<gg, 128, 0, stream>>>(nrm, wob, out, MR, DM, DM);
}
